// GCNModel_58128087384914
// MI455X (gfx1250) — compile-verified, matrix operand order checked by reading, not run on hardware
//
#include <hip/hip_runtime.h>
#include <hip/hip_bf16.h>
#include <cmath>

// ---------------------------------------------------------------------------
// Types for CDNA5 WMMA (wave32, 16x16x32 f16 -> f32)
// ---------------------------------------------------------------------------
typedef __attribute__((ext_vector_type(16))) _Float16 v16h;
typedef __attribute__((ext_vector_type(8)))  _Float16 v8h;
typedef __attribute__((ext_vector_type(8)))  float    v8f;

#define NEG_SLOPE 0.2f

// grid-stride loop over long totals
#define GSL(t, total)                                                          \
  for (long long t = blockIdx.x * (long long)blockDim.x + threadIdx.x;         \
       t < (total); t += (long long)gridDim.x * blockDim.x)

static inline int nblk(long long total, int bs) {
  long long b = (total + bs - 1) / bs;
  if (b > 1048576) b = 1048576;
  if (b < 1) b = 1;
  return (int)b;
}

// ---------------------------------------------------------------------------
// Elementwise helpers
// ---------------------------------------------------------------------------
__global__ void k_f32_to_f16(const float* __restrict__ in, _Float16* __restrict__ out,
                             long long n) {
  GSL(t, n) out[t] = (_Float16)in[t];
}

__global__ void k_fill(float* __restrict__ p, float v, long long n) {
  GSL(t, n) p[t] = v;
}

// ---------------------------------------------------------------------------
// Weight packing: W [K x Nc] f32 row-major -> per-lane WMMA B fragments (f16)
// fragment index: ((nt*Kt + kt)*32 + lane)*16 + e
// lane holds column nt*16 + (lane&15); koff = (lane>>4)*8
// e in 0..7  -> K = kt*32 + koff + e
// e in 8..15 -> K = kt*32 + 16 + koff + (e-8)
// ---------------------------------------------------------------------------
__global__ void k_pack_w(const float* __restrict__ W, _Float16* __restrict__ P,
                         int K, int Nc) {
  int Kt = K >> 5;
  int tilesN = Nc >> 4;
  int total = tilesN * Kt * 32;
  int t = blockIdx.x * blockDim.x + threadIdx.x;
  if (t >= total) return;
  int lane = t & 31;
  int kt = (t >> 5) % Kt;
  int nt = (t >> 5) / Kt;
  int col  = nt * 16 + (lane & 15);
  int koff = (lane >> 4) << 3;
  _Float16* dst = P + (size_t)t * 16;
#pragma unroll
  for (int e = 0; e < 8; ++e) {
    int k0 = kt * 32 + koff + e;
    int k1 = kt * 32 + 16 + koff + e;
    dst[e]     = (_Float16)W[(size_t)k0 * Nc + col];
    dst[e + 8] = (_Float16)W[(size_t)k1 * Nc + col];
  }
}

// ---------------------------------------------------------------------------
// WMMA GEMM, 2x2 register-blocked: each wave computes a 32x32 C super-tile
// (4 WMMA tiles) so every A/B fragment load feeds two v_wmma ops
// (2 b128 loads per WMMA instead of 4 -> half the L2 traffic).
// M: any multiple of 16 (odd tile count handled: clamped compute, guarded
// store -- wave-uniform, EXEC stays all-ones for WMMA). Nc/16 must be even.
// K multiple of 32.
// ---------------------------------------------------------------------------
__global__ void k_gemm_wmma(const _Float16* __restrict__ A,
                            const _Float16* __restrict__ Bp,
                            float* __restrict__ C,
                            int M, int Nc, int K) {
  int lane = threadIdx.x & 31;
  long long wave = (blockIdx.x * (long long)blockDim.x + threadIdx.x) >> 5;
  int Mt = M >> 4, Nt = Nc >> 4;
  int Mst = (Mt + 1) >> 1, Nst = Nt >> 1;
  long long stiles = (long long)Mst * Nst;
  if (wave >= stiles) return;            // wave-uniform: EXEC stays all-ones
  int smt = (int)(wave / Nst);
  int snt = (int)(wave % Nst);
  int mt0 = smt * 2, mt1 = smt * 2 + 1;
  int nt0 = snt * 2, nt1 = snt * 2 + 1;
  bool hasM1 = (mt1 < Mt);
  int mt1c = hasM1 ? mt1 : mt0;          // clamp for compute; store guarded
  int Kt = K >> 5;

  int r    = lane & 15;
  int koff = (lane >> 4) << 3;
  const _Float16* a0row = A + (size_t)(mt0 * 16 + r) * K;
  const _Float16* a1row = A + (size_t)(mt1c * 16 + r) * K;
  const v16h* b0 = (const v16h*)Bp + (size_t)nt0 * Kt * 32 + lane;
  const v16h* b1 = (const v16h*)Bp + (size_t)nt1 * Kt * 32 + lane;

  v8f acc00 = {}, acc01 = {}, acc10 = {}, acc11 = {};
  for (int kt = 0; kt < Kt; ++kt) {
    int kb = kt << 5;
    v8h a0lo = *(const v8h*)(a0row + kb + koff);
    v8h a0hi = *(const v8h*)(a0row + kb + 16 + koff);
    v8h a1lo = *(const v8h*)(a1row + kb + koff);
    v8h a1hi = *(const v8h*)(a1row + kb + 16 + koff);
    v16h a0, a1;
#pragma unroll
    for (int i = 0; i < 8; ++i) {
      a0[i] = a0lo[i]; a0[i + 8] = a0hi[i];
      a1[i] = a1lo[i]; a1[i + 8] = a1hi[i];
    }
    v16h bb0 = b0[(size_t)kt * 32];
    v16h bb1 = b1[(size_t)kt * 32];
    acc00 = __builtin_amdgcn_wmma_f32_16x16x32_f16(false, a0, false, bb0, (short)0, acc00, false, false);
    acc01 = __builtin_amdgcn_wmma_f32_16x16x32_f16(false, a0, false, bb1, (short)0, acc01, false, false);
    acc10 = __builtin_amdgcn_wmma_f32_16x16x32_f16(false, a1, false, bb0, (short)0, acc10, false, false);
    acc11 = __builtin_amdgcn_wmma_f32_16x16x32_f16(false, a1, false, bb1, (short)0, acc11, false, false);
  }

  int crow0 = mt0 * 16 + ((lane >> 4) << 3);
  int crow1 = mt1 * 16 + ((lane >> 4) << 3);
  int ccol0 = nt0 * 16 + (lane & 15);
  int ccol1 = nt1 * 16 + (lane & 15);
#pragma unroll
  for (int rr = 0; rr < 8; ++rr) {
    C[(size_t)(crow0 + rr) * Nc + ccol0] = acc00[rr];
    C[(size_t)(crow0 + rr) * Nc + ccol1] = acc01[rr];
  }
  if (hasM1) {
#pragma unroll
    for (int rr = 0; rr < 8; ++rr) {
      C[(size_t)(crow1 + rr) * Nc + ccol0] = acc10[rr];
      C[(size_t)(crow1 + rr) * Nc + ccol1] = acc11[rr];
    }
  }
}

// ---------------------------------------------------------------------------
// GAT attention scores per (node, head) + init max/sum accumulators
// ---------------------------------------------------------------------------
__global__ void k_attn(const float* __restrict__ Hf, const float* __restrict__ as_,
                       const float* __restrict__ ad_, float* __restrict__ es,
                       float* __restrict__ ed, float* __restrict__ mx,
                       float* __restrict__ sm, int n, int H, int C) {
  GSL(t, (long long)n * H) {
    int node = (int)(t / H), h = (int)(t % H);
    const float* hp = Hf + (size_t)node * H * C + (size_t)h * C;
    const float* ap = as_ + h * C;
    const float* dp = ad_ + h * C;
    float ss = 0.f, sd = 0.f;
    for (int c = 0; c < C; ++c) { float v = hp[c]; ss += v * ap[c]; sd += v * dp[c]; }
    es[t] = ss; ed[t] = sd;
    mx[t] = -INFINITY; sm[t] = 0.f;
  }
}

__device__ inline void edge_pair(const int* __restrict__ ei, long long e, int E,
                                 int& s, int& d) {
  if (e < E) { s = ei[e]; d = ei[(long long)E + e]; }
  else       { s = d = (int)(e - E); }        // self loops appended
}

__device__ inline void atomicMaxFloat(float* addr, float val) {
  if (val >= 0.f) atomicMax((int*)addr, __float_as_int(val));
  else            atomicMin((unsigned int*)addr, (unsigned int)__float_as_int(val));
}

__global__ void k_edge_max(const int* __restrict__ ei, const float* __restrict__ es,
                           const float* __restrict__ ed, float* __restrict__ mx,
                           int E, long long ET, int H) {
  GSL(t, ET * H) {
    long long e = t / H; int h = (int)(t % H);
    int s, d; edge_pair(ei, e, E, s, d);
    float v = es[(size_t)s * H + h] + ed[(size_t)d * H + h];
    v = v >= 0.f ? v : NEG_SLOPE * v;
    atomicMaxFloat(&mx[(size_t)d * H + h], v);
  }
}

__global__ void k_edge_sum(const int* __restrict__ ei, const float* __restrict__ es,
                           const float* __restrict__ ed, const float* __restrict__ mx,
                           float* __restrict__ sm, int E, long long ET, int H) {
  GSL(t, ET * H) {
    long long e = t / H; int h = (int)(t % H);
    int s, d; edge_pair(ei, e, E, s, d);
    float v = es[(size_t)s * H + h] + ed[(size_t)d * H + h];
    v = v >= 0.f ? v : NEG_SLOPE * v;
    atomicAdd(&sm[(size_t)d * H + h], __expf(v - mx[(size_t)d * H + h]));
  }
}

// One thread per (edge, channel): 256/512 consecutive lanes cover one edge's
// feature row -> coalesced gathers and same-cacheline atomic bursts.
__global__ void k_edge_gat(const int* __restrict__ ei, const float* __restrict__ es,
                           const float* __restrict__ ed, const float* __restrict__ mx,
                           const float* __restrict__ sm, const float* __restrict__ Hf,
                           float* __restrict__ agg, int E, long long ET, int H, int C) {
  int HC = H * C;
  GSL(t, ET * HC) {
    long long e = t / HC;
    int hc = (int)(t % HC);
    int h = hc / C;
    int s, d; edge_pair(ei, e, E, s, d);
    size_t di = (size_t)d * H + h;
    float v = es[(size_t)s * H + h] + ed[di];
    v = v >= 0.f ? v : NEG_SLOPE * v;
    float w = __expf(v - mx[di]) / (sm[di] + 1e-16f);
    atomicAdd(&agg[(size_t)d * HC + hc], Hf[(size_t)s * HC + hc] * w);
  }
}

// ---------------------------------------------------------------------------
// GCN degree / norm / aggregate
// ---------------------------------------------------------------------------
__global__ void k_deg(const int* __restrict__ ei, float* __restrict__ deg,
                      int E, long long ET) {
  GSL(t, ET) {
    int s, d; edge_pair(ei, t, E, s, d);
    atomicAdd(&deg[d], 1.0f);
  }
}

__global__ void k_dis(const float* __restrict__ deg, float* __restrict__ dis, int n) {
  GSL(t, (long long)n) {
    float g = deg[t];
    dis[t] = g > 0.f ? rsqrtf(g) : 0.f;
  }
}

__global__ void k_edge_gcn(const int* __restrict__ ei, const float* __restrict__ dis,
                           const float* __restrict__ Hf, float* __restrict__ agg,
                           int E, long long ET, int C) {
  GSL(t, ET * C) {
    long long e = t / C; int c = (int)(t % C);
    int s, d; edge_pair(ei, e, E, s, d);
    float nrm = dis[s] * dis[d];
    atomicAdd(&agg[(size_t)d * C + c], Hf[(size_t)s * C + c] * nrm);
  }
}

// ---------------------------------------------------------------------------
// pre-LN combines: agg + residual + two biases
// ---------------------------------------------------------------------------
__global__ void k_add3(const float* __restrict__ agg, const float* __restrict__ res,
                       const float* __restrict__ b1, const float* __restrict__ b2,
                       float* __restrict__ pre, int D, long long total) {
  GSL(t, total) {
    int c = (int)(t % D);
    pre[t] = agg[t] + res[t] + b1[c] + b2[c];
  }
}

// layer-3 special: mean over 8 heads of 64 + biases + residual
__global__ void k_pre3(const float* __restrict__ agg3, const float* __restrict__ r3,
                       const float* __restrict__ gb, const float* __restrict__ mb,
                       float* __restrict__ pre, int n) {
  GSL(t, (long long)n * 64) {
    int node = (int)(t / 64), c = (int)(t % 64);
    const float* a = agg3 + (size_t)node * 512;
    float s = 0.f;
#pragma unroll
    for (int h = 0; h < 8; ++h) s += a[h * 64 + c];
    pre[t] = s * 0.125f + gb[c] + r3[t] + mb[c];
  }
}

// ---------------------------------------------------------------------------
// LayerNorm: one block per row (blockDim == D), outputs f32 and/or f16
// ---------------------------------------------------------------------------
__global__ void k_ln(const float* __restrict__ in, const float* __restrict__ w,
                     const float* __restrict__ b, float* __restrict__ outF,
                     _Float16* __restrict__ outH, int D) {
  extern __shared__ float sh[];
  int row = blockIdx.x, tid = threadIdx.x;
  float v = in[(size_t)row * D + tid];
  sh[tid] = v; __syncthreads();
  for (int s = D >> 1; s > 0; s >>= 1) { if (tid < s) sh[tid] += sh[tid + s]; __syncthreads(); }
  float mu = sh[0] / D; __syncthreads();
  float dv = (v - mu) * (v - mu);
  sh[tid] = dv; __syncthreads();
  for (int s = D >> 1; s > 0; s >>= 1) { if (tid < s) sh[tid] += sh[tid + s]; __syncthreads(); }
  float var = sh[0] / D;
  float o = (v - mu) * rsqrtf(var + 1e-5f) * w[tid] + b[tid];
  if (outF) outF[(size_t)row * D + tid] = o;
  if (outH) outH[(size_t)row * D + tid] = (_Float16)o;
}

// ---------------------------------------------------------------------------
// Final thin layer (64 -> 2): plain VALU, too thin for WMMA
// ---------------------------------------------------------------------------
__global__ void k_l4(const float* __restrict__ x3, const float* __restrict__ gW,
                     const float* __restrict__ mW, float* __restrict__ h4,
                     float* __restrict__ r4, int n) {
  GSL(t, (long long)n) {
    const float* xr = x3 + (size_t)t * 64;
#pragma unroll
    for (int j = 0; j < 2; ++j) {
      float a = 0.f, r = 0.f;
      for (int k = 0; k < 64; ++k) { float xv = xr[k]; a += xv * gW[k * 2 + j]; r += xv * mW[k * 2 + j]; }
      h4[(size_t)t * 2 + j] = a;
      r4[(size_t)t * 2 + j] = r;
    }
  }
}

__global__ void k_final(const float* __restrict__ agg4, const float* __restrict__ r4,
                        const float* __restrict__ gb, const float* __restrict__ mb,
                        float* __restrict__ out, long long total) {
  GSL(t, total) {
    int c = (int)(t & 1);
    out[t] = agg4[t] + gb[c] + r4[t] + mb[c];
  }
}

// ---------------------------------------------------------------------------
// Launcher
// ---------------------------------------------------------------------------
extern "C" void kernel_launch(void* const* d_in, const int* in_sizes, int n_in,
                              void* d_out, int out_size, void* d_ws, size_t ws_size,
                              hipStream_t stream) {
  const float* x     = (const float*)d_in[0];
  const int*   ei    = (const int*)d_in[1];
  const float* g1_W  = (const float*)d_in[2];
  const float* g1_as = (const float*)d_in[3];
  const float* g1_ad = (const float*)d_in[4];
  const float* g1_b  = (const float*)d_in[5];
  const float* g2_W  = (const float*)d_in[6];
  const float* g2_b  = (const float*)d_in[7];
  const float* g3_W  = (const float*)d_in[8];
  const float* g3_as = (const float*)d_in[9];
  const float* g3_ad = (const float*)d_in[10];
  const float* g3_b  = (const float*)d_in[11];
  const float* g4_W  = (const float*)d_in[12];
  const float* g4_b  = (const float*)d_in[13];
  const float* ln0w  = (const float*)d_in[14];
  const float* ln0b  = (const float*)d_in[15];
  const float* ln1w  = (const float*)d_in[16];
  const float* ln1b  = (const float*)d_in[17];
  const float* ln2w  = (const float*)d_in[18];
  const float* ln2b  = (const float*)d_in[19];
  const float* m1_W  = (const float*)d_in[20];
  const float* m1_b  = (const float*)d_in[21];
  const float* m2_W  = (const float*)d_in[22];
  const float* m2_b  = (const float*)d_in[23];
  const float* m3_W  = (const float*)d_in[24];
  const float* m3_b  = (const float*)d_in[25];
  const float* m4_W  = (const float*)d_in[26];
  const float* m4_b  = (const float*)d_in[27];

  const int N  = in_sizes[0] / 1024;       // 50000
  const int E  = in_sizes[1] / 2;          // 800000
  const long long ET = (long long)E + N;   // + self loops

  // ---- workspace layout (lifetime-aliased) ----
  char* ws = (char*)d_ws;
  size_t off = 0;
  auto alloc = [&](size_t bytes) { size_t o = off; off += (bytes + 255) & ~(size_t)255; return o; };

  const size_t XH_BYTES  = (size_t)N * 1024 * 2;   // 102.4 MB
  const size_t R12_BYTES = (size_t)N * 512 * 4;    // 102.4 MB
  const size_t AGA_BYTES = (size_t)N * 256 * 4;    //  51.2 MB

  size_t oXH  = alloc(XH_BYTES);   // xh (L1) -> agg2 (L2) -> h3/pre3 (L3)
  size_t oR12 = alloc(R12_BYTES);  // h1 | r1 (L1), pre1, agg3 (L3)
  size_t oAGA = alloc(AGA_BYTES);  // agg1 (L1) -> h2 | r2 (L2), pre2
  size_t oX1H = alloc((size_t)N * 256 * 2);
  size_t oX2H = alloc((size_t)N * 128 * 2);
  size_t oX3  = alloc((size_t)N * 64 * 4);
  size_t oR3  = alloc((size_t)N * 64 * 4);
  size_t oES  = alloc((size_t)N * 8 * 4);
  size_t oED  = alloc((size_t)N * 8 * 4);
  size_t oMX  = alloc((size_t)N * 8 * 4);
  size_t oSM  = alloc((size_t)N * 8 * 4);
  size_t oDEG = alloc((size_t)N * 4);
  size_t oDIS = alloc((size_t)N * 4);
  size_t oH4  = alloc((size_t)N * 2 * 4);
  size_t oR4  = alloc((size_t)N * 2 * 4);
  size_t oA4  = alloc((size_t)N * 2 * 4);
  size_t oPG1 = alloc((size_t)1024 * 256 * 2);
  size_t oPM1 = alloc((size_t)1024 * 256 * 2);
  size_t oPG2 = alloc((size_t)256 * 128 * 2);
  size_t oPM2 = alloc((size_t)256 * 128 * 2);
  size_t oPG3 = alloc((size_t)128 * 512 * 2);
  size_t oPM3 = alloc((size_t)128 * 64 * 2);
  (void)ws_size;

  _Float16* xh   = (_Float16*)(ws + oXH);
  float*    agg2 = (float*)(ws + oXH);
  float*    h3   = (float*)(ws + oXH);
  float*    pre3 = (float*)(ws + oXH);
  float*    h1   = (float*)(ws + oR12);
  float*    r1   = (float*)(ws + oR12 + (size_t)N * 256 * 4);
  float*    pre1 = h1;
  float*    agg3 = (float*)(ws + oR12);
  float*    agg1 = (float*)(ws + oAGA);
  float*    h2   = (float*)(ws + oAGA);
  float*    r2   = (float*)(ws + oAGA + (size_t)N * 128 * 4);
  float*    pre2 = h2;
  _Float16* x1h  = (_Float16*)(ws + oX1H);
  _Float16* x2h  = (_Float16*)(ws + oX2H);
  float*    x3   = (float*)(ws + oX3);
  float*    r3   = (float*)(ws + oR3);
  float*    es   = (float*)(ws + oES);
  float*    ed   = (float*)(ws + oED);
  float*    mx   = (float*)(ws + oMX);
  float*    sm   = (float*)(ws + oSM);
  float*    deg  = (float*)(ws + oDEG);
  float*    dis  = (float*)(ws + oDIS);
  float*    h4   = (float*)(ws + oH4);
  float*    r4   = (float*)(ws + oR4);
  float*    agg4 = (float*)(ws + oA4);
  _Float16* pg1  = (_Float16*)(ws + oPG1);
  _Float16* pm1  = (_Float16*)(ws + oPM1);
  _Float16* pg2  = (_Float16*)(ws + oPG2);
  _Float16* pm2  = (_Float16*)(ws + oPM2);
  _Float16* pg3  = (_Float16*)(ws + oPG3);
  _Float16* pm3  = (_Float16*)(ws + oPM3);

  const int BS = 256;
  auto gemm = [&](const _Float16* A, const _Float16* Bp, float* C, int M, int Nc, int K) {
    long long stiles = (long long)((M / 16 + 1) / 2) * ((Nc / 16) / 2);
    long long thr = stiles * 32;
    hipLaunchKernelGGL(k_gemm_wmma, dim3((unsigned)((thr + BS - 1) / BS)), dim3(BS), 0, stream,
                       A, Bp, C, M, Nc, K);
  };
  auto pack = [&](const float* W, _Float16* P, int K, int Nc) {
    int total = (Nc / 16) * (K / 32) * 32;
    hipLaunchKernelGGL(k_pack_w, dim3((total + BS - 1) / BS), dim3(BS), 0, stream, W, P, K, Nc);
  };

  // ---- phase 0: conversions, weight packing, degree norm ----
  hipLaunchKernelGGL(k_f32_to_f16, dim3(nblk((long long)N * 1024, BS)), dim3(BS), 0, stream,
                     x, xh, (long long)N * 1024);
  pack(g1_W, pg1, 1024, 256);
  pack(m1_W, pm1, 1024, 256);
  pack(g2_W, pg2, 256, 128);
  pack(m2_W, pm2, 256, 128);
  pack(g3_W, pg3, 128, 512);
  pack(m3_W, pm3, 128, 64);
  hipLaunchKernelGGL(k_fill, dim3(nblk(N, BS)), dim3(BS), 0, stream, deg, 0.f, (long long)N);
  hipLaunchKernelGGL(k_deg, dim3(nblk(ET, BS)), dim3(BS), 0, stream, ei, deg, E, ET);
  hipLaunchKernelGGL(k_dis, dim3(nblk(N, BS)), dim3(BS), 0, stream, deg, dis, N);

  // ---- layer 1: GAT(1024 -> 8x32 concat) + MLP residual + LN(256) ----
  gemm(xh, pg1, h1, N, 256, 1024);
  gemm(xh, pm1, r1, N, 256, 1024);
  hipLaunchKernelGGL(k_attn, dim3(nblk((long long)N * 8, BS)), dim3(BS), 0, stream,
                     h1, g1_as, g1_ad, es, ed, mx, sm, N, 8, 32);
  hipLaunchKernelGGL(k_fill, dim3(nblk((long long)N * 256, BS)), dim3(BS), 0, stream,
                     agg1, 0.f, (long long)N * 256);
  hipLaunchKernelGGL(k_edge_max, dim3(nblk(ET * 8, BS)), dim3(BS), 0, stream, ei, es, ed, mx, E, ET, 8);
  hipLaunchKernelGGL(k_edge_sum, dim3(nblk(ET * 8, BS)), dim3(BS), 0, stream, ei, es, ed, mx, sm, E, ET, 8);
  hipLaunchKernelGGL(k_edge_gat, dim3(nblk(ET * 256, BS)), dim3(BS), 0, stream,
                     ei, es, ed, mx, sm, h1, agg1, E, ET, 8, 32);
  hipLaunchKernelGGL(k_add3, dim3(nblk((long long)N * 256, BS)), dim3(BS), 0, stream,
                     agg1, r1, g1_b, m1_b, pre1, 256, (long long)N * 256);
  hipLaunchKernelGGL(k_ln, dim3(N), dim3(256), 256 * sizeof(float), stream,
                     pre1, ln0w, ln0b, (float*)nullptr, x1h, 256);

  // ---- layer 2: GCN(256 -> 128) + MLP residual + LN(128) ----
  gemm(x1h, pg2, h2, N, 128, 256);
  gemm(x1h, pm2, r2, N, 128, 256);
  hipLaunchKernelGGL(k_fill, dim3(nblk((long long)N * 128, BS)), dim3(BS), 0, stream,
                     agg2, 0.f, (long long)N * 128);
  hipLaunchKernelGGL(k_edge_gcn, dim3(nblk(ET * 128, BS)), dim3(BS), 0, stream,
                     ei, dis, h2, agg2, E, ET, 128);
  hipLaunchKernelGGL(k_add3, dim3(nblk((long long)N * 128, BS)), dim3(BS), 0, stream,
                     agg2, r2, g2_b, m2_b, pre2, 128, (long long)N * 128);
  hipLaunchKernelGGL(k_ln, dim3(N), dim3(128), 128 * sizeof(float), stream,
                     pre2, ln1w, ln1b, (float*)nullptr, x2h, 128);

  // ---- layer 3: GAT(128 -> 8x64, mean heads) + MLP residual + LN(64) ----
  gemm(x2h, pg3, h3, N, 512, 128);
  gemm(x2h, pm3, r3, N, 64, 128);
  hipLaunchKernelGGL(k_attn, dim3(nblk((long long)N * 8, BS)), dim3(BS), 0, stream,
                     h3, g3_as, g3_ad, es, ed, mx, sm, N, 8, 64);
  hipLaunchKernelGGL(k_fill, dim3(nblk((long long)N * 512, BS)), dim3(BS), 0, stream,
                     agg3, 0.f, (long long)N * 512);
  hipLaunchKernelGGL(k_edge_max, dim3(nblk(ET * 8, BS)), dim3(BS), 0, stream, ei, es, ed, mx, E, ET, 8);
  hipLaunchKernelGGL(k_edge_sum, dim3(nblk(ET * 8, BS)), dim3(BS), 0, stream, ei, es, ed, mx, sm, E, ET, 8);
  hipLaunchKernelGGL(k_edge_gat, dim3(nblk(ET * 512, BS)), dim3(BS), 0, stream,
                     ei, es, ed, mx, sm, h3, agg3, E, ET, 8, 64);
  hipLaunchKernelGGL(k_pre3, dim3(nblk((long long)N * 64, BS)), dim3(BS), 0, stream,
                     agg3, r3, g3_b, m3_b, pre3, N);
  hipLaunchKernelGGL(k_ln, dim3(N), dim3(64), 64 * sizeof(float), stream,
                     pre3, ln2w, ln2b, x3, (_Float16*)nullptr, 64);

  // ---- layer 4: GCN(64 -> 2) + MLP residual ----
  hipLaunchKernelGGL(k_l4, dim3(nblk(N, BS)), dim3(BS), 0, stream, x3, g4_W, m4_W, h4, r4, N);
  hipLaunchKernelGGL(k_fill, dim3(nblk((long long)N * 2, BS)), dim3(BS), 0, stream,
                     agg4, 0.f, (long long)N * 2);
  hipLaunchKernelGGL(k_edge_gcn, dim3(nblk(ET * 2, BS)), dim3(BS), 0, stream,
                     ei, dis, h4, agg4, E, ET, 2);
  hipLaunchKernelGGL(k_final, dim3(nblk((long long)N * 2, BS)), dim3(BS), 0, stream,
                     agg4, r4, g4_b, m4_b, (float*)d_out, (long long)N * 2);
}
